// GAT_GCN_49847390437804
// MI455X (gfx1250) — matrix operand order repaired and checked by reading, not compiled, not run on hardware
//
#include <hip/hip_runtime.h>
#include <cstddef>

// ---------------------------------------------------------------------------
// GAT_GCN full forward for MI455X (gfx1250, wave32, WMMA bf16 -> f32 accum).
// Heavy GEMMs (transformer layers, LSTM input projections, FC heads) run on
// v_wmma_f32_16x16x32_bf16 with 64x64 block tiles (2 WMMA tiles per wave per
// K-chunk). LSTM recurrence keeps whh + h + c in LDS (320KB/WGP) and does the
// serial per-step 16x512x128 matmul with WMMA. Graph scatters use f32 global
// atomics. LDS tiles are laid out so every WMMA K-pair is one aligned dword
// (8x ds_load_b32 per fragment, conflict-free strides).
// ---------------------------------------------------------------------------

#define NN      3840        // nodes
#define NPG     30          // nodes per graph
#define BG      128         // graphs / batch
#define NE      30720       // edges (before self loops)
#define HEADS   10
#define DXD     78
#define HD      780         // HEADS*DXD

typedef __attribute__((ext_vector_type(16))) __bf16 bf16x16;
typedef __attribute__((ext_vector_type(8)))  float  f32x8;

union FragU { unsigned u[8]; bf16x16 v; };

__device__ __forceinline__ __bf16 f2bf(float f) {
  unsigned u = __float_as_uint(f);
  unsigned r = (u + 0x7FFFu + ((u >> 16) & 1u)) >> 16;
  unsigned short s = (unsigned short)r;
  __bf16 h;
  __builtin_memcpy(&h, &s, sizeof(h));
  return h;
}

__device__ __forceinline__ float sigf(float x) { return 1.f / (1.f + expf(-x)); }

// ---------------------------------------------------------------------------
// Generic GEMM: C[M,N] = act(scale * (A[M,K] @ B) + bias)
//   transB=1: B is W[N,K] row-major with leading dim ldb (PyTorch x@W.T)
//   transB=0: B is W[K,N] row-major with leading dim ldb
// act: 0=none 1=relu 2=sigmoid
// 256 threads = 8 waves; block tile 64x64; wave (mi,ni) owns rows
// {mi*16..+15, mi*16+32..+47} x cols {ni*16..+15}: 1 B-frag, 2 A-frags,
// 2 WMMAs per 32-wide K chunk.
// ---------------------------------------------------------------------------
__launch_bounds__(256)
__global__ void k_gemm(const float* __restrict__ A, int lda,
                       const float* __restrict__ B, int ldb, int transB,
                       const float* __restrict__ bias,
                       float* __restrict__ C, int ldc,
                       int M, int N, int K, float scale, int act)
{
  __shared__ __bf16 As[64][36];      // pitch 36 -> 72B rows, 4B aligned, 18-dword
                                     // stride => conflict-free fragment reads
  __shared__ __bf16 Bs[16][64][2];   // k-pair major: [k>>1][n][k&1]
  const int tid  = threadIdx.x;
  const int lane = tid & 31;
  const int wave = tid >> 5;
  const int m0 = blockIdx.y * 64;
  const int n0 = blockIdx.x * 64;
  const int mi = wave >> 2;          // 0..1
  const int ni = wave & 3;           // 0..3
  const int lr = lane & 15;
  const int hh = lane >> 4;

  f32x8 acc0 = {0.f, 0.f, 0.f, 0.f, 0.f, 0.f, 0.f, 0.f};
  f32x8 acc1 = {0.f, 0.f, 0.f, 0.f, 0.f, 0.f, 0.f, 0.f};

  for (int k0 = 0; k0 < K; k0 += 32) {
    // ---- stage A tile (64x32): fast path is block-uniform (no EXEC churn)
    if (m0 + 64 <= M && k0 + 32 <= K) {
      #pragma unroll
      for (int i = 0; i < 8; ++i) {
        int lin = tid + i * 256;
        int r = lin >> 5, k = lin & 31;
        As[r][k] = f2bf(A[(size_t)(m0 + r) * lda + k0 + k]);
      }
    } else {
      #pragma unroll
      for (int i = 0; i < 8; ++i) {
        int lin = tid + i * 256;
        int r = lin >> 5, k = lin & 31;
        int gm = m0 + r, gk = k0 + k;
        float v = (gm < M && gk < K) ? A[(size_t)gm * lda + gk] : 0.f;
        As[r][k] = f2bf(v);
      }
    }
    // ---- stage B tile (32x64) into k-pair-major layout
    if (transB) {
      if (n0 + 64 <= N && k0 + 32 <= K) {
        #pragma unroll
        for (int i = 0; i < 8; ++i) {
          int lin = tid + i * 256;
          int n = lin >> 5, k = lin & 31;          // contiguous over k
          Bs[k >> 1][n][k & 1] = f2bf(B[(size_t)(n0 + n) * ldb + k0 + k]);
        }
      } else {
        #pragma unroll
        for (int i = 0; i < 8; ++i) {
          int lin = tid + i * 256;
          int n = lin >> 5, k = lin & 31;
          int gn = n0 + n, gk = k0 + k;
          float v = (gn < N && gk < K) ? B[(size_t)gn * ldb + gk] : 0.f;
          Bs[k >> 1][n][k & 1] = f2bf(v);
        }
      }
    } else {
      if (n0 + 64 <= N && k0 + 32 <= K) {
        #pragma unroll
        for (int i = 0; i < 8; ++i) {
          int lin = tid + i * 256;
          int k = lin >> 6, n = lin & 63;          // contiguous over n
          Bs[k >> 1][n][k & 1] = f2bf(B[(size_t)(k0 + k) * ldb + n0 + n]);
        }
      } else {
        #pragma unroll
        for (int i = 0; i < 8; ++i) {
          int lin = tid + i * 256;
          int k = lin >> 6, n = lin & 63;
          int gn = n0 + n, gk = k0 + k;
          float v = (gn < N && gk < K) ? B[(size_t)gk * ldb + gn] : 0.f;
          Bs[k >> 1][n][k & 1] = f2bf(v);
        }
      }
    }
    __syncthreads();

    // ---- fragments via 32-bit LDS reads (each dword = one aligned K pair)
    FragU a0, a1, bfr;
    const unsigned* Ar0 = (const unsigned*)&As[mi * 16 + lr][0];
    const unsigned* Ar1 = (const unsigned*)&As[mi * 16 + 32 + lr][0];
    const unsigned* Bp  = (const unsigned*)&Bs[0][0][0];
    #pragma unroll
    for (int i = 0; i < 8; ++i) {
      // A (16x32, 16-bit): lane m=lr, K halves interleave per VGPR pair
      int di = (i < 4) ? (hh * 4 + i) : (8 + hh * 4 + (i - 4));
      a0.u[i] = Ar0[di];
      a1.u[i] = Ar1[di];
      // B (32x16, 16-bit): lane n=lr, same K interleave as A, 2 K per VGPR
      bfr.u[i] = Bp[di * 64 + ni * 16 + lr];
    }
    acc0 = __builtin_amdgcn_wmma_f32_16x16x32_bf16(false, a0.v, false, bfr.v,
                                                   (short)0, acc0, false, false);
    acc1 = __builtin_amdgcn_wmma_f32_16x16x32_bf16(false, a1.v, false, bfr.v,
                                                   (short)0, acc1, false, false);
    __syncthreads();
  }

  const int cn = n0 + ni * 16 + lr;
  const float bv = (bias && cn < N) ? bias[cn] : 0.f;
  #pragma unroll
  for (int r = 0; r < 8; ++r) {
    int cm0 = m0 + mi * 16 + hh * 8 + r;
    int cm1 = cm0 + 32;
    if (cm0 < M && cn < N) {
      float v = acc0[r] * scale + bv;
      if (act == 1) v = fmaxf(v, 0.f);
      else if (act == 2) v = sigf(v);
      C[(size_t)cm0 * ldc + cn] = v;
    }
    if (cm1 < M && cn < N) {
      float v = acc1[r] * scale + bv;
      if (act == 1) v = fmaxf(v, 0.f);
      else if (act == 2) v = sigf(v);
      C[(size_t)cm1 * ldc + cn] = v;
    }
  }
}

// ---------------------------------------------------------------------------
// Residual LayerNorm: y = LN(x + o) * g + b   (one row per block)
// ---------------------------------------------------------------------------
__launch_bounds__(256)
__global__ void k_ln(const float* __restrict__ x, const float* __restrict__ o,
                     const float* __restrict__ g, const float* __restrict__ b,
                     float* __restrict__ y, int D)
{
  __shared__ float red[256];
  const int row = blockIdx.x, tid = threadIdx.x;
  const float* xr = x + (size_t)row * D;
  const float* orr = o + (size_t)row * D;
  float s = 0.f;
  for (int i = tid; i < D; i += 256) s += xr[i] + orr[i];
  red[tid] = s; __syncthreads();
  for (int st = 128; st > 0; st >>= 1) { if (tid < st) red[tid] += red[tid + st]; __syncthreads(); }
  const float mean = red[0] / D;
  __syncthreads();
  float v = 0.f;
  for (int i = tid; i < D; i += 256) { float d = xr[i] + orr[i] - mean; v += d * d; }
  red[tid] = v; __syncthreads();
  for (int st = 128; st > 0; st >>= 1) { if (tid < st) red[tid] += red[tid + st]; __syncthreads(); }
  const float inv = rsqrtf(red[0] / D + 1e-5f);
  for (int i = tid; i < D; i += 256)
    y[(size_t)row * D + i] = (xr[i] + orr[i] - mean) * inv * g[i] + b[i];
}

// Row softmax in place (one row per block)
__launch_bounds__(256)
__global__ void k_softmax(float* __restrict__ s, int D)
{
  __shared__ float red[256];
  const int row = blockIdx.x, tid = threadIdx.x;
  float* r = s + (size_t)row * D;
  float mx = -3.4e38f;
  for (int i = tid; i < D; i += 256) mx = fmaxf(mx, r[i]);
  red[tid] = mx; __syncthreads();
  for (int st = 128; st > 0; st >>= 1) { if (tid < st) red[tid] = fmaxf(red[tid], red[tid + st]); __syncthreads(); }
  mx = red[0]; __syncthreads();
  float sum = 0.f;
  for (int i = tid; i < D; i += 256) { float e = expf(r[i] - mx); r[i] = e; sum += e; }
  red[tid] = sum; __syncthreads();
  for (int st = 128; st > 0; st >>= 1) { if (tid < st) red[tid] += red[tid + st]; __syncthreads(); }
  const float inv = 1.f / red[0];
  for (int i = tid; i < D; i += 256) r[i] *= inv;
}

// ---------------------------------------------------------------------------
// LSTM recurrence: per step g = xp[t] + h @ whh.T ; gates ; write h.
// One block per 16-row batch tile; whh cached in LDS as bf16.
// xp: [128,T,512] f32; out: [128,T,256] at column dirofs.
// whh pitch 130 (260B rows) / h pitch 132 (264B rows): even pitches keep every
// K pair a 4B-aligned dword; pair strides 65 / 66 dwords are conflict-free.
// ---------------------------------------------------------------------------
#define SHM_RECUR (512*130*2 + 16*132*2 + 16*512*4 + 16*128*4)
__launch_bounds__(256)
__global__ void k_lstm_recur(const float* __restrict__ xp,
                             const float* __restrict__ whh,   // [512][128] f32
                             float* __restrict__ out,
                             int T, int dirofs, int reverse)
{
  extern __shared__ char smem[];
  __bf16* whh_s = (__bf16*)smem;                               // 512 x (128+2)
  __bf16* h_s   = (__bf16*)(smem + 512 * 130 * 2);             // 16 x (128+4)
  float*  g_s   = (float*)(smem + 512 * 130 * 2 + 16 * 132 * 2);
  float*  c_s   = g_s + 16 * 512;
  const int tid = threadIdx.x, lane = tid & 31, wave = tid >> 5;
  const int lr = lane & 15, hh = lane >> 4;
  const int b0 = blockIdx.x * 16;

  for (int i = 0; i < 256; ++i) {                 // 65536 weights, coalesced
    int idx = i * 256 + tid;
    int n = idx >> 7, k = idx & 127;
    whh_s[n * 130 + k] = f2bf(whh[n * 128 + k]);
  }
  for (int i = tid; i < 16 * 132; i += 256) h_s[i] = f2bf(0.f);
  for (int i = tid; i < 16 * 128; i += 256) c_s[i] = 0.f;
  __syncthreads();

  for (int s = 0; s < T; ++s) {
    const int t = reverse ? (T - 1 - s) : s;
    // 8 waves x 4 tiles cover N=512
    for (int nt4 = 0; nt4 < 4; ++nt4) {
      const int nbase = (wave * 4 + nt4) * 16;
      f32x8 acc;
      #pragma unroll
      for (int r = 0; r < 8; ++r) {
        int m = hh * 8 + r;
        acc[r] = xp[(size_t)((b0 + m) * T + t) * 512 + nbase + lr];
      }
      const unsigned* Hrow = (const unsigned*)&h_s[lr * 132];
      const unsigned* Wrow = (const unsigned*)&whh_s[(nbase + lr) * 130];
      #pragma unroll
      for (int kc = 0; kc < 4; ++kc) {
        FragU af, bfr;
        #pragma unroll
        for (int i = 0; i < 8; ++i) {
          int di = (i < 4) ? (hh * 4 + i) : (8 + hh * 4 + (i - 4));
          af.u[i]  = Hrow[kc * 16 + di];
          bfr.u[i] = Wrow[kc * 16 + di];
        }
        acc = __builtin_amdgcn_wmma_f32_16x16x32_bf16(false, af.v, false, bfr.v,
                                                      (short)0, acc, false, false);
      }
      #pragma unroll
      for (int r = 0; r < 8; ++r) {
        int m = hh * 8 + r;
        g_s[m * 512 + nbase + lr] = acc[r];
      }
    }
    __syncthreads();
    // gates: i,f,g,o at offsets 0,128,256,384
    #pragma unroll
    for (int rep = 0; rep < 8; ++rep) {
      int cell = rep * 256 + tid;
      int m = cell >> 7, n = cell & 127;
      float gi = g_s[m * 512 + n];
      float gf = g_s[m * 512 + 128 + n];
      float gg = g_s[m * 512 + 256 + n];
      float go = g_s[m * 512 + 384 + n];
      float c = sigf(gf) * c_s[m * 128 + n] + sigf(gi) * tanhf(gg);
      float hv = sigf(go) * tanhf(c);
      c_s[m * 128 + n] = c;
      h_s[m * 132 + n] = f2bf(hv);
      out[(size_t)((b0 + m) * T + t) * 256 + dirofs + n] = hv;
    }
    __syncthreads();
  }
}

// ---------------------------------------------------------------------------
// Small elementwise / scatter kernels
// ---------------------------------------------------------------------------
__global__ void k_fill(float* p, float v, int n)
{ int i = blockIdx.x * 256 + threadIdx.x; if (i < n) p[i] = v; }

__global__ void k_embed(const int* __restrict__ tok, const float* __restrict__ emb,
                        float* __restrict__ out, int total, int C)
{
  int i = blockIdx.x * 256 + threadIdx.x;
  if (i >= total * C) return;
  int c = i % C, r = i / C;
  out[i] = emb[(size_t)tok[r] * C + c];
}

__global__ void k_bias_act(float* __restrict__ x, const float* __restrict__ bias,
                           int rows, int C, int act)
{
  int i = blockIdx.x * 256 + threadIdx.x;
  if (i >= rows * C) return;
  float v = x[i] + bias[i % C];
  if (act == 1) v = fmaxf(v, 0.f);
  x[i] = v;
}

__global__ void k_mp3_relu(const float* __restrict__ x, float* __restrict__ y,
                           int rows, int Cin)
{
  int Cout = Cin / 3;
  int i = blockIdx.x * 256 + threadIdx.x;
  if (i >= rows * Cout) return;
  int j = i % Cout, r = i / Cout;
  const float* xr = x + (size_t)r * Cin + 3 * j;
  y[i] = fmaxf(fmaxf(fmaxf(xr[0], xr[1]), xr[2]), 0.f);
}

__global__ void k_conv1d(const float* __restrict__ x, const float* __restrict__ w,
                         const float* __restrict__ bias, float* __restrict__ y,
                         int Bn, int Cin, int Lin, int Cout, int KW)
{
  int Lout = Lin - KW + 1;
  int i = blockIdx.x * 256 + threadIdx.x;
  if (i >= Bn * Cout * Lout) return;
  int p = i % Lout, rest = i / Lout;
  int o = rest % Cout, bb = rest / Cout;
  float acc = bias[o];
  for (int c = 0; c < Cin; ++c) {
    const float* xr = x + ((size_t)bb * Cin + c) * Lin + p;
    const float* wr = w + ((size_t)o * Cin + c) * KW;
    for (int kk = 0; kk < KW; ++kk) acc += xr[kk] * wr[kk];
  }
  y[i] = acc;
}

__device__ __forceinline__ void atomicMaxF(float* addr, float val)
{
  unsigned* ua = (unsigned*)addr;
  unsigned old = *ua;
  while (__uint_as_float(old) < val) {
    unsigned assumed = old;
    old = atomicCAS(ua, assumed, __float_as_uint(val));
    if (old == assumed) break;
  }
}

__global__ void k_gat_coef(const float* __restrict__ xw, const float* __restrict__ asrc,
                           const float* __restrict__ adst,
                           float* __restrict__ as_, float* __restrict__ ad_)
{
  int i = blockIdx.x * 256 + threadIdx.x;
  if (i >= NN * HEADS) return;
  int h = i % HEADS, n = i / HEADS;
  const float* xr = xw + (size_t)n * HD + h * DXD;
  float s1 = 0.f, s2 = 0.f;
  for (int d = 0; d < DXD; ++d) { s1 += xr[d] * asrc[h * DXD + d]; s2 += xr[d] * adst[h * DXD + d]; }
  as_[i] = s1; ad_[i] = s2;
}

__device__ __forceinline__ void edge_sd(const int* ei, int e, int& src, int& dst)
{
  if (e < NE) { src = ei[e]; dst = ei[NE + e]; }
  else        { src = e - NE; dst = e - NE; }
}

__global__ void k_gat_p1(const int* __restrict__ ei, const float* __restrict__ as_,
                         const float* __restrict__ ad_, float* __restrict__ m)
{
  int i = blockIdx.x * 256 + threadIdx.x;
  if (i >= (NE + NN) * HEADS) return;
  int e = i / HEADS, h = i % HEADS, src, dst;
  edge_sd(ei, e, src, dst);
  float a = as_[src * HEADS + h] + ad_[dst * HEADS + h];
  a = (a > 0.f) ? a : 0.2f * a;
  atomicMaxF(&m[dst * HEADS + h], a);
}

__global__ void k_gat_p2(const int* __restrict__ ei, const float* __restrict__ as_,
                         const float* __restrict__ ad_, const float* __restrict__ m,
                         float* __restrict__ s)
{
  int i = blockIdx.x * 256 + threadIdx.x;
  if (i >= (NE + NN) * HEADS) return;
  int e = i / HEADS, h = i % HEADS, src, dst;
  edge_sd(ei, e, src, dst);
  float a = as_[src * HEADS + h] + ad_[dst * HEADS + h];
  a = (a > 0.f) ? a : 0.2f * a;
  atomicAdd(&s[dst * HEADS + h], expf(a - m[dst * HEADS + h]));
}

__global__ void k_gat_p3(const int* __restrict__ ei, const float* __restrict__ as_,
                         const float* __restrict__ ad_, const float* __restrict__ m,
                         const float* __restrict__ s, const float* __restrict__ xw,
                         float* __restrict__ out)
{
  int i = blockIdx.x * 256 + threadIdx.x;
  if (i >= (NE + NN) * HD) return;
  int e = i / HD, c = i % HD, h = c / DXD, src, dst;
  edge_sd(ei, e, src, dst);
  float a = as_[src * HEADS + h] + ad_[dst * HEADS + h];
  a = (a > 0.f) ? a : 0.2f * a;
  float ev = expf(a - m[dst * HEADS + h]);
  float alpha = ev / (s[dst * HEADS + h] + 1e-16f);
  atomicAdd(&out[(size_t)dst * HD + c], alpha * xw[(size_t)src * HD + c]);
}

__global__ void k_gcn_deg(const int* __restrict__ ei, float* __restrict__ deg)
{
  int e = blockIdx.x * 256 + threadIdx.x;
  if (e >= NE + NN) return;
  int src, dst; edge_sd(ei, e, src, dst);
  atomicAdd(&deg[dst], 1.f);
}

__global__ void k_dinv(const float* __restrict__ deg, float* __restrict__ dinv)
{
  int i = blockIdx.x * 256 + threadIdx.x;
  if (i < NN) dinv[i] = (deg[i] > 0.f) ? rsqrtf(deg[i]) : 0.f;
}

__global__ void k_gcn_scatter(const int* __restrict__ ei, const float* __restrict__ dinv,
                              const float* __restrict__ xw, float* __restrict__ out)
{
  int i = blockIdx.x * 256 + threadIdx.x;
  if (i >= (NE + NN) * HD) return;
  int e = i / HD, c = i % HD, src, dst;
  edge_sd(ei, e, src, dst);
  float norm = dinv[src] * dinv[dst];
  atomicAdd(&out[(size_t)dst * HD + c], norm * xw[(size_t)src * HD + c]);
}

// graph pooling: per-graph max + mean over the 30 contiguous nodes
__global__ void k_pool(const float* __restrict__ h, float* __restrict__ g)
{
  int gi = blockIdx.x;
  for (int c = threadIdx.x; c < HD; c += blockDim.x) {
    float mx = -3.4e38f, sm = 0.f;
    for (int n = 0; n < NPG; ++n) {
      float v = h[(size_t)(gi * NPG + n) * HD + c];
      mx = fmaxf(mx, v); sm += v;
    }
    g[gi * (2 * HD) + c] = mx;
    g[gi * (2 * HD) + HD + c] = sm / (float)NPG;
  }
}

__global__ void k_copy_xg(const float* __restrict__ xc, float* __restrict__ outp)
{
  int i = blockIdx.x * 256 + threadIdx.x;
  if (i >= BG * 128) return;
  outp[128 + i] = xc[(i / 128) * 512 + (i % 128)];
}

// ---------------------------------------------------------------------------
// Host-side orchestration
// ---------------------------------------------------------------------------
extern "C" void kernel_launch(void* const* d_in, const int* in_sizes, int n_in,
                              void* d_out, int out_size, void* d_ws, size_t ws_size,
                              hipStream_t stream)
{
  (void)in_sizes; (void)n_in; (void)out_size; (void)ws_size;
  float* WS = (float*)d_ws;
  auto F = [&](int i) { return (const float*)d_in[i]; };

  // ---- workspace offsets (floats). Graph phase (<=44.5M) is overlaid by the
  // sequence phase buffers after it completes (stream-ordered).
  const size_t OFF_QKV = 0;            // 3840*2340
  const size_t OFF_SC  = 9000000;      // 3840*3840
  const size_t OFF_T1  = 23800000;     // 3840*2048
  const size_t OFF_T2  = 31700000;     // 3840*780
  const size_t OFF_HA  = 34700000;     // 3840*780
  const size_t OFF_HB  = 37700000;     // 3840*780
  const size_t OFF_AS  = 40700000;     // 3840*10 each:
  const size_t OFF_AD  = OFF_AS + 38400;
  const size_t OFF_M   = OFF_AD + 38400;
  const size_t OFF_S   = OFF_M + 38400;
  const size_t OFF_DEG = OFF_S + 38400;
  const size_t OFF_DIV = OFF_DEG + 3840;
  const size_t OFF_GC  = 41000000;     // 128*1560
  const size_t OFF_LN  = 41400000;     // 3840*780
  const size_t OFF_SQA = 0;            // 128*735*256 (overlay)
  const size_t OFF_SQB = 24100000;     // 128*735*256
  const size_t OFF_XP  = 48200000;     // 128*735*512
  const size_t OFF_XC  = 96400000;     // 128*512 (persists)
  const size_t OFF_FC2 = 96500000;     // 128*1024
  const size_t OFF_FC3 = 96700000;     // 128*128
  const size_t OFF_CVA = 96800000;     // 128*167*256
  const size_t OFF_CVB = 102300000;    // conv scratch

  float* QKV = WS + OFF_QKV;  float* SC  = WS + OFF_SC;
  float* T1  = WS + OFF_T1;   float* T2  = WS + OFF_T2;
  float* HA  = WS + OFF_HA;   float* HB  = WS + OFF_HB;
  float* ASB = WS + OFF_AS;   float* ADB = WS + OFF_AD;
  float* MB  = WS + OFF_M;    float* SB  = WS + OFF_S;
  float* DEG = WS + OFF_DEG;  float* DIV = WS + OFF_DIV;
  float* GC  = WS + OFF_GC;   float* LNB = WS + OFF_LN;
  float* SQA = WS + OFF_SQA;  float* SQB = WS + OFF_SQB;
  float* XP  = WS + OFF_XP;   float* XC  = WS + OFF_XC;
  float* FC2 = WS + OFF_FC2;  float* FC3 = WS + OFF_FC3;
  float* CVA = WS + OFF_CVA;  float* CVB = WS + OFF_CVB;

  auto gemm = [&](const float* A, int lda, const float* B, int ldb, int tB,
                  const float* bias, float* C, int ldc, int M, int N, int K,
                  float scale, int act) {
    dim3 g((N + 63) / 64, (M + 63) / 64);
    k_gemm<<<g, 256, 0, stream>>>(A, lda, B, ldb, tB, bias, C, ldc, M, N, K, scale, act);
  };
  auto NB = [](int n) { return (n + 255) / 256; };

  // ---- transformer layer (in X -> out O), params at pbase
  auto run_tf = [&](const float* X, float* O, int d, int pbase) {
    int d3 = 3 * d;
    gemm(X, d, F(pbase + 0), d, 1, F(pbase + 1), QKV, d3, NN, d3, d, 1.f, 0);  // qkv
    gemm(QKV, d3, QKV + d, d3, 1, nullptr, SC, NN, NN, NN, d,
         rsqrtf((float)d), 0);                                                 // q@k.T/sqrt(d)
    k_softmax<<<NN, 256, 0, stream>>>(SC, NN);
    gemm(SC, NN, QKV + 2 * d, d3, 0, nullptr, T1, d, NN, d, NN, 1.f, 0);       // attn@v
    gemm(T1, d, F(pbase + 2), d, 1, F(pbase + 3), T2, d, NN, d, d, 1.f, 0);    // @wout+bout
    k_ln<<<NN, 256, 0, stream>>>(X, T2, F(pbase + 4), F(pbase + 5), LNB, d);   // ln1
    gemm(LNB, d, F(pbase + 6), d, 1, F(pbase + 7), T1, 2048, NN, 2048, d, 1.f, 1);
    gemm(T1, 2048, F(pbase + 8), 2048, 1, F(pbase + 9), T2, d, NN, d, 2048, 1.f, 0);
    k_ln<<<NN, 256, 0, stream>>>(LNB, T2, F(pbase + 10), F(pbase + 11), O, d); // ln2
  };

  // ---- one 2-layer bi-LSTM stack followed by relu+maxpool3 (256 -> 85)
  auto run_stack = [&](float*& cur, float*& nxt, int T, int in0, int pbase) {
    for (int l = 0; l < 2; ++l) {
      int ind = (l == 0) ? in0 : 256;
      int pb = pbase + l * 6;
      gemm(cur, ind, F(pb + 0), ind, 1, F(pb + 2), XP, 512, BG * T, 512, ind, 1.f, 0);
      k_lstm_recur<<<8, 256, SHM_RECUR, stream>>>(XP, F(pb + 1), nxt, T, 0, 0);
      gemm(cur, ind, F(pb + 3), ind, 1, F(pb + 5), XP, 512, BG * T, 512, ind, 1.f, 0);
      k_lstm_recur<<<8, 256, SHM_RECUR, stream>>>(XP, F(pb + 4), nxt, T, 128, 1);
      float* t = cur; cur = nxt; nxt = t;
    }
    k_mp3_relu<<<NB(BG * T * 85), 256, 0, stream>>>(cur, nxt, BG * T, 256);
    float* t = cur; cur = nxt; nxt = t;
  };

  // ---- sequence branch: embed -> 3 stacks -> fc -> xc column
  auto run_seq = [&](const int* tok, int T, const float* emb,
                     int l1, int l2, int l3, int fcb, float* xc_col) {
    float* cur = SQA; float* nxt = SQB;
    k_embed<<<NB(BG * T * 256), 256, 0, stream>>>(tok, emb, cur, BG * T, 256);
    run_stack(cur, nxt, T, 256, l1);
    run_stack(cur, nxt, T, 85, l2);
    run_stack(cur, nxt, T, 85, l3);
    int Kf = T * 85;
    gemm(cur, Kf, F(fcb + 0), Kf, 1, F(fcb + 1), FC2, 1024, BG, 1024, Kf, 1.f, 1);
    gemm(FC2, 1024, F(fcb + 2), 1024, 1, F(fcb + 3), xc_col, 512, BG, 128, 1024, 1.f, 0);
  };

  const int* ei = (const int*)d_in[1];

  // ================= graph phase =================
  run_tf(F(0), HA, 78, 6);                                        // tf1
  // GAT (input HA, output HB relu'd)
  gemm(HA, 78, F(30), HD, 0, nullptr, T1, HD, NN, HD, 78, 1.f, 0); // xw
  k_gat_coef<<<NB(NN * HEADS), 256, 0, stream>>>(T1, F(31), F(32), ASB, ADB);
  k_fill<<<NB(NN * HEADS), 256, 0, stream>>>(MB, -3.0e38f, NN * HEADS);
  k_fill<<<NB(NN * HEADS), 256, 0, stream>>>(SB, 0.f, NN * HEADS);
  k_fill<<<NB(NN * HD), 256, 0, stream>>>(HB, 0.f, NN * HD);
  k_gat_p1<<<NB((NE + NN) * HEADS), 256, 0, stream>>>(ei, ASB, ADB, MB);
  k_gat_p2<<<NB((NE + NN) * HEADS), 256, 0, stream>>>(ei, ASB, ADB, MB, SB);
  k_gat_p3<<<NB((NE + NN) * HD), 256, 0, stream>>>(ei, ASB, ADB, MB, SB, T1, HB);
  k_bias_act<<<NB(NN * HD), 256, 0, stream>>>(HB, F(33), NN, HD, 1);
  run_tf(HB, HA, HD, 18);                                         // tf2
  // GCN (input HA, output HB relu'd)
  gemm(HA, HD, F(34), HD, 0, nullptr, T1, HD, NN, HD, HD, 1.f, 0);
  k_fill<<<NB(NN), 256, 0, stream>>>(DEG, 0.f, NN);
  k_gcn_deg<<<NB(NE + NN), 256, 0, stream>>>(ei, DEG);
  k_dinv<<<NB(NN), 256, 0, stream>>>(DEG, DIV);
  k_fill<<<NB(NN * HD), 256, 0, stream>>>(HB, 0.f, NN * HD);
  k_gcn_scatter<<<NB((NE + NN) * HD), 256, 0, stream>>>(ei, DIV, T1, HB);
  k_bias_act<<<NB(NN * HD), 256, 0, stream>>>(HB, F(35), NN, HD, 1);
  // pooling + fcg -> xg into xc[:,0:128]
  k_pool<<<BG, 256, 0, stream>>>(HB, GC);
  gemm(GC, 1560, F(36), 1560, 1, F(37), FC2, 1024, BG, 1024, 1560, 1.f, 1);
  gemm(FC2, 1024, F(38), 1024, 1, F(39), XC + 0, 512, BG, 128, 1024, 1.f, 0);

  // ================= sequence branches (overlay graph buffers) =================
  run_seq((const int*)d_in[3], 735, F(40), 41, 53, 65, 77, XC + 384);   // target -> xc[:,384:512]
  run_seq((const int*)d_in[5], 100, F(92), 93, 105, 117, 129, XC + 128);// drug   -> xc[:,128:256]

  // ================= fingerprint conv branch =================
  k_embed<<<NB(BG * 167 * 256), 256, 0, stream>>>((const int*)d_in[4], F(81), CVA, BG * 167, 256);
  k_conv1d<<<NB(BG * 32 * 249), 256, 0, stream>>>(CVA, F(82), F(83), CVB, BG, 167, 256, 32, 8);
  k_mp3_relu<<<NB(BG * 32 * 83), 256, 0, stream>>>(CVB, CVA, BG * 32, 249);
  k_conv1d<<<NB(BG * 64 * 76), 256, 0, stream>>>(CVA, F(84), F(85), CVB, BG, 32, 83, 64, 8);
  k_mp3_relu<<<NB(BG * 64 * 25), 256, 0, stream>>>(CVB, CVA, BG * 64, 76);
  k_conv1d<<<NB(BG * 128 * 18), 256, 0, stream>>>(CVA, F(86), F(87), CVB, BG, 64, 25, 128, 8);
  k_mp3_relu<<<NB(BG * 128 * 6), 256, 0, stream>>>(CVB, CVA, BG * 128, 18);
  gemm(CVA, 768, F(88), 768, 1, F(89), FC2, 1024, BG, 1024, 768, 1.f, 1);
  gemm(FC2, 1024, F(90), 1024, 1, F(91), XC + 256, 512, BG, 128, 1024, 1.f, 0);

  // ================= head =================
  gemm(XC, 512, F(133), 512, 1, F(134), FC2, 1024, BG, 1024, 512, 1.f, 1);
  gemm(FC2, 1024, F(135), 1024, 1, F(136), FC3, 128, BG, 128, 1024, 1.f, 1);
  gemm(FC3, 128, F(137), 128, 1, F(138), (float*)d_out, 1, BG, 1, 128, 1.f, 2); // sigmoid out
  k_copy_xg<<<NB(BG * 128), 256, 0, stream>>>(XC, (float*)d_out);               // xg output
}
